// SelfRoutingCapsuleLayer_36283883716617
// MI455X (gfx1250) — matrix operand order repaired and checked by reading, not compiled, not run on hardware
//
#include <hip/hip_runtime.h>
#include <hip/hip_bf16.h>

// ---------------------------------------------------------------------------
// Capsule layer: Conv2D(3x3,128->256,SAME,ReLU) via bf16 WMMA implicit GEMM
// with async global->LDS double-buffered staging, then 3 rounds of dynamic
// routing, then broadcast v over HxW.
// ---------------------------------------------------------------------------

#define B_   16
#define H_   64
#define W_   64
#define CIN  128
#define COUT 256
#define KSZ  1152        // 3*3*128
#define NPOS 4096        // H_*W_
#define LDS_STRIDE 40    // 32 bf16 + 8 pad per w-row

typedef __attribute__((ext_vector_type(16))) __bf16 v16bf;
typedef __attribute__((ext_vector_type(8)))  float  v8f;

union FragBF { uint4 q[2]; v16bf v; };

static __device__ __forceinline__ unsigned short f2bf(float f) {
  // round-to-nearest-even f32 -> bf16
  unsigned int u = __float_as_uint(f);
  return (unsigned short)((u + 0x7FFFu + ((u >> 16) & 1u)) >> 16);
}

// ---------------- workspace layout (bytes, 256B aligned) -------------------
#define OFF_XBF   ((size_t)0)          // 16*64*64*128 bf16  = 16,777,216 B
#define OFF_WB    ((size_t)16777216)   // 256*1152 bf16      =    589,824 B
#define OFF_XCONV ((size_t)17367040)   // 16*4096*256 f32    = 67,108,864 B
#define OFF_BLOG  ((size_t)84475904)   // 16*4096*16 f32     =  4,194,304 B
#define OFF_S     ((size_t)88670208)   // 16*256 f32         =     16,384 B
#define OFF_V     ((size_t)88686592)   // 16*256 f32         =     16,384 B

// ---------------------------------------------------------------------------
__global__ void k_cvt_input(const float* __restrict__ in,
                            unsigned short* __restrict__ out, int n4) {
  int g = blockIdx.x * blockDim.x + threadIdx.x;
  if (g >= n4) return;
  float4 f = reinterpret_cast<const float4*>(in)[g];
  unsigned int lo = (unsigned int)f2bf(f.x) | ((unsigned int)f2bf(f.y) << 16);
  unsigned int hi = (unsigned int)f2bf(f.z) | ((unsigned int)f2bf(f.w) << 16);
  reinterpret_cast<uint2*>(out)[g] = make_uint2(lo, hi);
}

// HWIO (k-major x co) f32  ->  N-major bf16 [co][k] so B-fragment K runs are
// contiguous per lane.
__global__ void k_cvt_weights(const float* __restrict__ kin,
                              unsigned short* __restrict__ wB) {
  int g = blockIdx.x * blockDim.x + threadIdx.x;
  if (g >= KSZ * COUT) return;
  int co = g & (COUT - 1);
  int kk = g >> 8;
  wB[(size_t)co * KSZ + kk] = f2bf(kin[g]);
}

__global__ void k_zero_f4(float* p, int n4) {
  int g = blockIdx.x * blockDim.x + threadIdx.x;
  if (g < n4) reinterpret_cast<float4*>(p)[g] = make_float4(0.f, 0.f, 0.f, 0.f);
}

// ---------------------------------------------------------------------------
// Stage one 64x32 bf16 A-slice for k-step `ks` into LDS buffer `ldsBase`.
// In-bounds lanes issue global_load_async_to_lds_b128 (ASYNCcnt-tracked, no
// VGPR round trip); SAME-padding lanes zero-fill via a normal LDS store
// (DScnt-tracked, drained by the step barrier).
// ---------------------------------------------------------------------------
static __device__ __forceinline__ void stage_tile(
    const unsigned short* __restrict__ xbf, unsigned short* ldsBase,
    int b, int h, int ks, int sw, int sc) {
  const int kh = ks / 12;
  const int kw = (ks / 4) % 3;
  const int cc = ks & 3;            // 32-channel chunk
  const int hh = h + kh - 1;
  const int ww = sw + kw - 1;
  unsigned short* dst = ldsBase + sw * LDS_STRIDE + sc;
  if (hh >= 0 && hh < H_ && ww >= 0 && ww < W_) {
    const unsigned short* src =
        xbf + ((((size_t)b * H_ + hh) * W_ + ww) * CIN + cc * 32 + sc);
    const unsigned ldsA = (unsigned)(size_t)dst;               // LDS byte addr
    const unsigned long long ga = (unsigned long long)(size_t)src;
    asm volatile("global_load_async_to_lds_b128 %0, %1, off"
                 :: "v"(ldsA), "v"(ga) : "memory");
  } else {
    *reinterpret_cast<uint4*>(dst) = make_uint4(0, 0, 0, 0);
  }
}

// ---------------------------------------------------------------------------
// Conv: one block per (b,h) row. Tile M=64 (w), N=256 (co), K-loop 36 x 32.
// 8 waves; wave owns N-tiles {wave, wave+8} x M-tiles {0..3} -> 8 WMMAs/step.
// Double-buffered LDS: step k+1 streams in asynchronously under step k's WMMAs.
// ---------------------------------------------------------------------------
__global__ void __launch_bounds__(256)
k_conv_wmma(const unsigned short* __restrict__ xbf,
            const unsigned short* __restrict__ wB,
            const float* __restrict__ bias,
            float* __restrict__ xconv) {
  __shared__ unsigned short lA[2][64][LDS_STRIDE];   // 2 x 5120 B

  const int tid  = threadIdx.x;
  const int wave = tid >> 5;
  const int lane = tid & 31;
  const int bh   = blockIdx.x;
  const int b    = bh >> 6;
  const int h    = bh & 63;

  v8f acc[4][2];
#pragma unroll
  for (int mi = 0; mi < 4; ++mi) {
    acc[mi][0] = (v8f){0.f, 0.f, 0.f, 0.f, 0.f, 0.f, 0.f, 0.f};
    acc[mi][1] = (v8f){0.f, 0.f, 0.f, 0.f, 0.f, 0.f, 0.f, 0.f};
  }

  // LDS staging role: thread -> (w, 8-channel slice)
  const int sw = tid >> 2;            // 0..63
  const int sc = (tid & 3) * 8;       // 0,8,16,24

  const int laneM  = lane & 15;
  const int kBaseA = (lane < 16) ? 0 : 8;   // A: K halves {0-7,16-23}/{8-15,24-31}
  const int kBaseB = (lane < 16) ? 0 : 8;   // B: K halves {0-7,16-23}/{8-15,24-31}

  const unsigned short* wRow0 = wB + (size_t)(wave * 16 + laneM) * KSZ + kBaseB;
  const unsigned short* wRow1 = wB + (size_t)((wave + 8) * 16 + laneM) * KSZ + kBaseB;

  // prologue: stage step 0, drain, publish
  stage_tile(xbf, &lA[0][0][0], b, h, 0, sw, sc);
  asm volatile("s_wait_asynccnt 0x0" ::: "memory");
  __syncthreads();

  for (int ks = 0; ks < 36; ++ks) {
    const int cur = ks & 1;

    // kick off next tile's async staging into the other buffer
    if (ks < 35)
      stage_tile(xbf, &lA[cur ^ 1][0][0], b, h, ks + 1, sw, sc);

    // ---- B fragments straight from L2-resident transposed weights ----
    FragBF bf0, bf1;
    {
      const unsigned short* p0 = wRow0 + ks * 32;
      bf0.q[0] = *reinterpret_cast<const uint4*>(p0);
      bf0.q[1] = *reinterpret_cast<const uint4*>(p0 + 16);
      const unsigned short* p1 = wRow1 + ks * 32;
      bf1.q[0] = *reinterpret_cast<const uint4*>(p1);
      bf1.q[1] = *reinterpret_cast<const uint4*>(p1 + 16);
      // keep next step's weight lines warm
      __builtin_prefetch(p0 + 32, 0, 1);
      __builtin_prefetch(p1 + 32, 0, 1);
    }

#pragma unroll
    for (int mi = 0; mi < 4; ++mi) {
      FragBF af;
      const unsigned short* rp = &lA[cur][mi * 16 + laneM][kBaseA];
      af.q[0] = *reinterpret_cast<const uint4*>(rp);
      af.q[1] = *reinterpret_cast<const uint4*>(rp + 16);
      acc[mi][0] = __builtin_amdgcn_wmma_f32_16x16x32_bf16(
          false, af.v, false, bf0.v, (short)0, acc[mi][0], false, false);
      acc[mi][1] = __builtin_amdgcn_wmma_f32_16x16x32_bf16(
          false, af.v, false, bf1.v, (short)0, acc[mi][1], false, false);
    }

    // next buffer fully landed + everyone done with this one
    asm volatile("s_wait_asynccnt 0x0" ::: "memory");
    __syncthreads();
  }

  // ---- epilogue: bias + ReLU, store as [b][n][co] f32 ----
  const int mHalf = (lane < 16) ? 0 : 8;
#pragma unroll
  for (int j = 0; j < 2; ++j) {
    const int co = (wave + 8 * j) * 16 + laneM;
    const float bv = bias[co];
#pragma unroll
    for (int mi = 0; mi < 4; ++mi) {
#pragma unroll
      for (int r = 0; r < 8; ++r) {
        const int m = mi * 16 + r + mHalf;         // C-frag: M = r + 8*(lane>=16)
        float val = acc[mi][j][r] + bv;
        val = fmaxf(val, 0.0f);
        xconv[(((size_t)b * NPOS) + h * 64 + m) * COUT + co] = val;
      }
    }
  }
}

// ---------------------------------------------------------------------------
// Routing step 1: s[b,cap,d] = sum_n softmax(blog[b,n,:])[cap] * x[b,n,cap,d]
// block = 16 n x 16 cap threads, 16 n-subiters; LDS-atomic then global-atomic.
// grid = B*16 blocks (each covers 256 n's).
// ---------------------------------------------------------------------------
__global__ void k_route_s(const float* __restrict__ xconv,
                          const float* __restrict__ blog,
                          float* __restrict__ s) {
  const int blk   = blockIdx.x;
  const int b     = blk >> 4;
  const int chunk = blk & 15;
  const int tid   = threadIdx.x;
  const int cap   = tid & 15;
  const int ni    = tid >> 4;

  __shared__ float sAcc[256];
  sAcc[tid] = 0.f;
  __syncthreads();

  float accd[16];
#pragma unroll
  for (int d = 0; d < 16; ++d) accd[d] = 0.f;

  for (int sub = 0; sub < 16; ++sub) {
    const int n = chunk * 256 + sub * 16 + ni;
    const float* bl = blog + ((size_t)b * NPOS + n) * 16;
    float bv[16], mx = -3.402823466e38f;
#pragma unroll
    for (int c = 0; c < 16; ++c) { bv[c] = bl[c]; mx = fmaxf(mx, bv[c]); }
    float sum = 0.f;
#pragma unroll
    for (int c = 0; c < 16; ++c) sum += __expf(bv[c] - mx);
    const float cmy = __expf(bv[cap] - mx) / sum;

    const float* xr = xconv + (((size_t)b * NPOS + n) * COUT) + cap * 16;
#pragma unroll
    for (int d = 0; d < 16; ++d) accd[d] = fmaf(cmy, xr[d], accd[d]);
  }
#pragma unroll
  for (int d = 0; d < 16; ++d) atomicAdd(&sAcc[cap * 16 + d], accd[d]);
  __syncthreads();
  atomicAdd(&s[b * 256 + tid], sAcc[tid]);
}

// l2-normalize s -> v   (block per batch; thread = (cap,d))
__global__ void k_route_v(const float* __restrict__ s, float* __restrict__ v) {
  const int b   = blockIdx.x;
  const int tid = threadIdx.x;
  const int cap = tid >> 4;
  const float* sr = s + b * 256 + cap * 16;
  float ss = 0.f;
#pragma unroll
  for (int d = 0; d < 16; ++d) { const float t = sr[d]; ss = fmaf(t, t, ss); }
  const float inv = rsqrtf(fmaxf(ss, 1e-12f));
  v[b * 256 + tid] = s[b * 256 + tid] * inv;
}

// blog[b,n,cap] += dot(x[b,n,cap,:], v[b,cap,:])
__global__ void k_route_u(const float* __restrict__ xconv,
                          const float* __restrict__ v,
                          float* __restrict__ blog) {
  const size_t g   = (size_t)blockIdx.x * 256 + threadIdx.x;  // (b,n,cap)
  const int   cap  = (int)(g & 15);
  const size_t bn  = g >> 4;
  const int   b    = (int)(bn >> 12);
  const float* xr = xconv + bn * COUT + cap * 16;
  const float* vr = v + b * 256 + cap * 16;
  float dot = 0.f;
#pragma unroll
  for (int d = 0; d < 16; ++d) dot = fmaf(xr[d], vr[d], dot);
  blog[g] += dot;
}

// out[b,h,w,:] = v[b,:]
__global__ void k_bcast(const float* __restrict__ v, float* __restrict__ out) {
  const size_t g = (size_t)blockIdx.x * 256 + threadIdx.x;   // float4 index
  const size_t e = g * 4;
  const int b  = (int)(e >> 20);        // 64*64*256 = 2^20 per batch
  const int cd = (int)(e & 255);
  const float4 val = *reinterpret_cast<const float4*>(v + b * 256 + cd);
  *reinterpret_cast<float4*>(out + e) = val;
}

// ---------------------------------------------------------------------------
extern "C" void kernel_launch(void* const* d_in, const int* in_sizes, int n_in,
                              void* d_out, int out_size, void* d_ws, size_t ws_size,
                              hipStream_t stream) {
  const float* inX    = (const float*)d_in[0];   // (16,64,64,128)
  const float* inK    = (const float*)d_in[1];   // (3,3,128,256)
  const float* inBias = (const float*)d_in[2];   // (256,)
  float* out = (float*)d_out;
  char*  ws  = (char*)d_ws;

  unsigned short* xbf   = (unsigned short*)(ws + OFF_XBF);
  unsigned short* wB    = (unsigned short*)(ws + OFF_WB);
  float*          xconv = (float*)(ws + OFF_XCONV);
  float*          blog  = (float*)(ws + OFF_BLOG);
  float*          sbuf  = (float*)(ws + OFF_S);
  float*          vbuf  = (float*)(ws + OFF_V);

  // precision/layout conversion
  k_cvt_input  <<<8192, 256, 0, stream>>>(inX, xbf, 2097152);
  k_cvt_weights<<<1152, 256, 0, stream>>>(inK, wB);
  k_zero_f4    <<<1024, 256, 0, stream>>>(blog, 262144);

  // conv via WMMA implicit GEMM
  k_conv_wmma<<<B_ * H_, 256, 0, stream>>>(xbf, wB, inBias, xconv);

  // 3 routing iterations (final b-update is unused by the reference -> skip)
  for (int it = 0; it < 3; ++it) {
    k_zero_f4 <<<4,    256, 0, stream>>>(sbuf, 1024);
    k_route_s <<<256,  256, 0, stream>>>(xconv, blog, sbuf);
    k_route_v <<<16,   256, 0, stream>>>(sbuf, vbuf);
    if (it < 2) k_route_u<<<4096, 256, 0, stream>>>(xconv, vbuf, blog);
  }

  k_bcast<<<16384, 256, 0, stream>>>(vbuf, out);
}
